// GraphSAGE_38628935860964
// MI455X (gfx1250) — compile-verified, matrix operand order checked by reading, not run on hardware
//
#include <hip/hip_runtime.h>
#include <hip/hip_bf16.h>
#include <math.h>

// ---------------------------------------------------------------------------
// GraphSAGE (2-layer) for MI455X / gfx1250.
//
// Roofline reasoning:
//  * mean-aggregate moves ~819 MB of gathered rows + ~819 MB of atomic adds
//    per layer, but the node-feature matrices (100000 x 128 f32 = 51 MB) fit
//    entirely in the 192 MB L2, so scatter/gather is L2-resident. Plain
//    global_atomic_add_f32 with default (RT) temporal hints is the right path.
//  * GEMMs are ~8.6 GFLOP total -> negligible on the matrix cores; use the
//    dense bf16 WMMA path (v_wmma_f32_16x16x32_bf16) with f32 accumulate.
//  * All WMMA operands (A and transposed-staged B) are fetched from LDS as
//    16-byte ds_load_b128 with a 272-byte row pitch (conflict-free banks).
// ---------------------------------------------------------------------------

typedef __attribute__((ext_vector_type(16))) __bf16 v16bf;
typedef __attribute__((ext_vector_type(8)))  __bf16 v8bf;
typedef __attribute__((ext_vector_type(8)))  float  v8f;

#define NNODES 100000
#define DIM    128
#define LPITCH (DIM + 8)   // LDS row pitch in bf16: 272B -> 4-bank lane shift
#define NCLS   40
#define NCLSP  48          // 40 padded to 3 WMMA n-tiles of 16

union frag16 { v16bf v; v8bf h[2]; };

// ---------------- degree: deg[dst] += 1 ------------------------------------
__global__ __launch_bounds__(256) void sage_deg_kernel(
    const long long* __restrict__ dst, float* __restrict__ deg, int E) {
  int e = blockIdx.x * 256 + threadIdx.x;
  if (e < E) atomicAdd(&deg[(int)dst[e]], 1.0f);
}

// ---------------- scatter-add: sum[dst] += feat[src] ------------------------
// One 32-lane group per edge; each lane moves a float4 (32*4 = 128 feats).
__global__ __launch_bounds__(256) void sage_scatter_kernel(
    const float* __restrict__ feat, const long long* __restrict__ src,
    const long long* __restrict__ dst, float* __restrict__ sum, int E) {
  long long idx = (long long)blockIdx.x * 256 + threadIdx.x;
  int e  = (int)(idx >> 5);
  int ch = (int)(idx & 31);
  if (e >= E) return;
  long long s = src[e];
  long long d = dst[e];
  float4 v = ((const float4*)(feat + s * DIM))[ch];
  float* op = sum + d * DIM + ch * 4;
  atomicAdd(op + 0, v.x);
  atomicAdd(op + 1, v.y);
  atomicAdd(op + 2, v.z);
  atomicAdd(op + 3, v.w);
}

// ---------------- normalize in place: agg = sum / max(deg,1) ---------------
__global__ __launch_bounds__(256) void sage_norm_kernel(
    float* __restrict__ sum, const float* __restrict__ deg, int n) {
  long long idx = (long long)blockIdx.x * 256 + threadIdx.x;
  int node = (int)(idx >> 5);
  int ch   = (int)(idx & 31);
  if (node >= n) return;
  float inv = 1.0f / fmaxf(deg[node], 1.0f);
  float4* p = (float4*)(sum + (size_t)node * DIM) + ch;
  float4 v = *p;
  v.x *= inv; v.y *= inv; v.z *= inv; v.w *= inv;
  *p = v;
}

// Convert 8 consecutive floats to a v8bf pack.
__device__ __forceinline__ v8bf cvt8(const float* __restrict__ p) {
  float4 a = *(const float4*)p;
  float4 b = *(const float4*)(p + 4);
  v8bf r = {(__bf16)a.x, (__bf16)a.y, (__bf16)a.z, (__bf16)a.w,
            (__bf16)b.x, (__bf16)b.y, (__bf16)b.z, (__bf16)b.w};
  return r;
}

// Build a 16x32 bf16 WMMA operand fragment (ISA 7.12.2 16-bit layout) from an
// LDS row: elements 0..7 <- K = kb + 8*half .. +7 ; 8..15 <- K = kb+16+8*half.
__device__ __forceinline__ v16bf load_frag(const __bf16* row, int kb, int hl) {
  frag16 f;
  f.h[0] = *(const v8bf*)&row[kb + hl * 8];
  f.h[1] = *(const v8bf*)&row[kb + 16 + hl * 8];
  return f.v;
}

// ---------------- GEMM1: h = relu(agg@W1l + x@W1r + b1) --------------------
// Block = 256 threads = 8 waves; wave w owns n-tile [16w, 16w+16).
__global__ __launch_bounds__(256) void sage_gemm1_kernel(
    const float* __restrict__ agg, const float* __restrict__ x,
    const float* __restrict__ W1l, const float* __restrict__ W1r,
    const float* __restrict__ b1, float* __restrict__ h) {
  __shared__ __bf16 sA[16][LPITCH];        // agg tile
  __shared__ __bf16 sX[16][LPITCH];        // x tile
  __shared__ __bf16 sWtl[DIM][LPITCH];     // W1_l transposed: [n][k]
  __shared__ __bf16 sWtr[DIM][LPITCH];     // W1_r transposed: [n][k]
  __shared__ float  sB1[DIM];

  const int tid  = threadIdx.x;
  const int row0 = blockIdx.x * 16;        // 100000 % 16 == 0, no guards

  // A/X tiles: one 8-element chunk per thread (16 rows x 16 chunks = 256).
  {
    int r  = tid >> 4;
    int c8 = (tid & 15) * 8;
    *(v8bf*)&sA[r][c8] = cvt8(&agg[(size_t)(row0 + r) * DIM + c8]);
    *(v8bf*)&sX[r][c8] = cvt8(&x[(size_t)(row0 + r) * DIM + c8]);
  }
  // Weights: coalesced float4 reads, transposed scatter into LDS [n][k].
  for (int f = tid * 8; f < DIM * DIM; f += 256 * 8) {
    int k  = f >> 7;        // row of W (K index)
    int n8 = f & 127;       // starting col (N index), multiple of 8
    v8bf wl = cvt8(&W1l[(size_t)k * DIM + n8]);
    v8bf wr = cvt8(&W1r[(size_t)k * DIM + n8]);
#pragma unroll
    for (int j = 0; j < 8; ++j) {
      sWtl[n8 + j][k] = wl[j];
      sWtr[n8 + j][k] = wr[j];
    }
  }
  if (tid < DIM) sB1[tid] = b1[tid];
  __syncthreads();

  const int wv = tid >> 5;                 // 0..7 -> n-tile
  const int lane = tid & 31;
  const int hl = lane >> 4, mn = lane & 15;
  const int n0 = wv * 16;

  const __bf16* rowA = sA[mn];
  const __bf16* rowX = sX[mn];
  const __bf16* rowL = sWtl[n0 + mn];
  const __bf16* rowR = sWtr[n0 + mn];

  v8f c = {};
#pragma unroll
  for (int kt = 0; kt < DIM / 32; ++kt) {
    int kb = kt * 32;
    v16bf aA = load_frag(rowA, kb, hl);
    v16bf aX = load_frag(rowX, kb, hl);
    v16bf bL = load_frag(rowL, kb, hl);
    v16bf bR = load_frag(rowR, kb, hl);
    c = __builtin_amdgcn_wmma_f32_16x16x32_bf16(false, aA, false, bL,
                                                (short)0, c, false, false);
    c = __builtin_amdgcn_wmma_f32_16x16x32_bf16(false, aX, false, bR,
                                                (short)0, c, false, false);
  }
  // C layout: lanes 0-15 N=lane M=r; lanes 16-31 N=lane-16 M=r+8
#pragma unroll
  for (int r = 0; r < 8; ++r) {
    int M = r + hl * 8;
    int N = n0 + mn;
    float v = c[r] + sB1[N];
    h[(size_t)(row0 + M) * DIM + N] = v > 0.0f ? v : 0.0f;
  }
}

// ---------------- GEMM2: out = agg2@W2l + h@W2r + b2 (logits) --------------
// Block = 96 threads = 3 waves covering 48 padded output cols.
__global__ __launch_bounds__(96) void sage_gemm2_kernel(
    const float* __restrict__ agg, const float* __restrict__ hfeat,
    const float* __restrict__ W2l, const float* __restrict__ W2r,
    const float* __restrict__ b2, float* __restrict__ out) {
  __shared__ __bf16 sA[16][LPITCH];
  __shared__ __bf16 sH[16][LPITCH];
  __shared__ __bf16 sWtl[NCLSP][LPITCH];   // W2_l transposed: [n][k]
  __shared__ __bf16 sWtr[NCLSP][LPITCH];
  __shared__ float  sB[NCLSP];

  const int tid  = threadIdx.x;
  const int row0 = blockIdx.x * 16;

  for (int ci = tid; ci < 256; ci += 96) { // 16 rows x 16 chunks of 8
    int r  = ci >> 4;
    int c8 = (ci & 15) * 8;
    *(v8bf*)&sA[r][c8] = cvt8(&agg[(size_t)(row0 + r) * DIM + c8]);
    *(v8bf*)&sH[r][c8] = cvt8(&hfeat[(size_t)(row0 + r) * DIM + c8]);
  }
  for (int i = tid; i < DIM * NCLSP; i += 96) {
    int k = i / NCLSP, n = i % NCLSP;
    sWtl[n][k] = (n < NCLS) ? (__bf16)W2l[(size_t)k * NCLS + n] : (__bf16)0.0f;
    sWtr[n][k] = (n < NCLS) ? (__bf16)W2r[(size_t)k * NCLS + n] : (__bf16)0.0f;
  }
  if (tid < NCLSP) sB[tid] = (tid < NCLS) ? b2[tid] : 0.0f;
  __syncthreads();

  const int wv = tid >> 5;                 // 0..2
  const int lane = tid & 31;
  const int hl = lane >> 4, mn = lane & 15;
  const int n0 = wv * 16;

  const __bf16* rowA = sA[mn];
  const __bf16* rowH = sH[mn];
  const __bf16* rowL = sWtl[n0 + mn];
  const __bf16* rowR = sWtr[n0 + mn];

  v8f c = {};
#pragma unroll
  for (int kt = 0; kt < DIM / 32; ++kt) {
    int kb = kt * 32;
    v16bf aA = load_frag(rowA, kb, hl);
    v16bf aH = load_frag(rowH, kb, hl);
    v16bf bL = load_frag(rowL, kb, hl);
    v16bf bR = load_frag(rowR, kb, hl);
    c = __builtin_amdgcn_wmma_f32_16x16x32_bf16(false, aA, false, bL,
                                                (short)0, c, false, false);
    c = __builtin_amdgcn_wmma_f32_16x16x32_bf16(false, aH, false, bR,
                                                (short)0, c, false, false);
  }
#pragma unroll
  for (int r = 0; r < 8; ++r) {
    int M = r + hl * 8;
    int N = n0 + mn;
    if (N < NCLS)
      out[(size_t)(row0 + M) * NCLS + N] = c[r] + sB[N];
  }
}

// ---------------- log_softmax over 40 cols, one wave32 per row -------------
__global__ __launch_bounds__(256) void sage_logsoftmax_kernel(
    float* __restrict__ out, int nrows) {
  int wid = blockIdx.x * (256 >> 5) + (threadIdx.x >> 5);
  if (wid >= nrows) return;
  int lane = threadIdx.x & 31;
  float* row = out + (size_t)wid * NCLS;
  float v0 = row[lane];                                        // cols 0..31
  float v1 = (lane < NCLS - 32) ? row[lane + 32] : -INFINITY;  // cols 32..39
  float m = fmaxf(v0, v1);
#pragma unroll
  for (int off = 16; off > 0; off >>= 1)
    m = fmaxf(m, __shfl_xor(m, off, 32));
  float s = __expf(v0 - m) + ((lane < NCLS - 32) ? __expf(v1 - m) : 0.0f);
#pragma unroll
  for (int off = 16; off > 0; off >>= 1)
    s += __shfl_xor(s, off, 32);
  float ls = __logf(s);
  row[lane] = v0 - m - ls;
  if (lane < NCLS - 32) row[lane + 32] = v1 - m - ls;
}

// ---------------------------------------------------------------------------
extern "C" void kernel_launch(void* const* d_in, const int* in_sizes, int n_in,
                              void* d_out, int out_size, void* d_ws, size_t ws_size,
                              hipStream_t stream) {
  const float*     x   = (const float*)d_in[0];
  const long long* ei  = (const long long*)d_in[1];   // int64 [2, E]
  const float*     W1l = (const float*)d_in[2];
  const float*     b1  = (const float*)d_in[3];
  const float*     W1r = (const float*)d_in[4];
  const float*     W2l = (const float*)d_in[5];
  const float*     b2  = (const float*)d_in[6];
  const float*     W2r = (const float*)d_in[7];
  float* out = (float*)d_out;

  const int N = NNODES;
  const int E = in_sizes[1] / 2;
  const long long* srcp = ei;
  const long long* dstp = ei + E;

  // Workspace layout (256B aligned):
  //   deg : N floats
  //   agg : N*128 floats (sum -> mean, reused for layer 2)
  //   h   : N*128 floats (hidden activations)
  char*  ws  = (char*)d_ws;
  size_t off = 0;
  float* deg = (float*)(ws + off); off += (((size_t)N * 4) + 255) & ~(size_t)255;
  float* agg = (float*)(ws + off); off += (size_t)N * DIM * 4;
  float* h   = (float*)(ws + off);

  const int blk = 256;
  const int gridE    = (E + blk - 1) / blk;                        // degree
  const int gridE32  = (int)(((long long)E * 32 + blk - 1) / blk); // scatter
  const int gridN32  = (int)(((long long)N * 32 + blk - 1) / blk); // normalize
  const int gridTile = N / 16;                                     // 6250

  // -------- layer 1 aggregation --------
  hipMemsetAsync(deg, 0, (size_t)N * 4, stream);
  hipMemsetAsync(agg, 0, (size_t)N * DIM * 4, stream);
  sage_deg_kernel<<<gridE, blk, 0, stream>>>(dstp, deg, E);
  sage_scatter_kernel<<<gridE32, blk, 0, stream>>>(x, srcp, dstp, agg, E);
  sage_norm_kernel<<<gridN32, blk, 0, stream>>>(agg, deg, N);

  // -------- layer 1 GEMM + bias + ReLU --------
  sage_gemm1_kernel<<<gridTile, 256, 0, stream>>>(agg, x, W1l, W1r, b1, h);

  // -------- layer 2 aggregation (reuse agg buffer) --------
  hipMemsetAsync(agg, 0, (size_t)N * DIM * 4, stream);
  sage_scatter_kernel<<<gridE32, blk, 0, stream>>>(h, srcp, dstp, agg, E);
  sage_norm_kernel<<<gridN32, blk, 0, stream>>>(agg, deg, N);

  // -------- layer 2 GEMM + bias, then log_softmax in place --------
  sage_gemm2_kernel<<<gridTile, 96, 0, stream>>>(agg, h, W2l, W2r, b2, out);
  sage_logsoftmax_kernel<<<(N + 7) / 8, 256, 0, stream>>>(out, N);
}
